// EpipolarTransformer_6073083757162
// MI455X (gfx1250) — matrix operand order repaired and checked by reading, not compiled, not run on hardware
//
#include <hip/hip_runtime.h>
#include <hip/hip_bf16.h>
#include <math.h>

#define DEV static __device__ __forceinline__

typedef __attribute__((ext_vector_type(16))) __bf16          v16bf;
typedef __attribute__((ext_vector_type(16))) unsigned short  v16u;
typedef __attribute__((ext_vector_type(8)))  float           v8f;
typedef __attribute__((ext_vector_type(4)))  unsigned        v4ui;
typedef __attribute__((ext_vector_type(8)))  int             v8si;
typedef __attribute__((ext_vector_type(4)))  int             v4si;

#if defined(__HIP_DEVICE_COMPILE__) && __has_builtin(__builtin_amdgcn_tensor_load_to_lds) && __has_builtin(__builtin_amdgcn_s_wait_tensorcnt)
#define USE_TDM 1
#else
#define USE_TDM 0
#endif

// ---- problem dims ----
constexpr int DB = 2, DV = 2, DC = 128, DHIN = 64, DWIN = 64;
constexpr int DH = 32, DW = 32, DR = 1024, DS = 96, DOCT = 10;
constexpr int DHEADS = 4, DDH = 32, DINNER = 128, DMLPD = 256;
constexpr int NNV = DB * DV;        // 4
constexpr int NT  = NNV * DR;       // 4096 tokens / rays
constexpr float FEPS = 1e-6f;

// ---- workspace layout (bytes) ----
constexpr size_t OFF_FEATS = 0;                                      // f32 NHWC 4x32x32x128 (also x tokens)
constexpr size_t SZ_FEATS  = (size_t)NT * DC * 4;
constexpr size_t OFF_Z     = OFF_FEATS + SZ_FEATS;                   // bf16 [nr][s][c]
constexpr size_t SZ_Z      = (size_t)NT * DS * DC * 2;
constexpr size_t OFF_UPF   = OFF_Z + SZ_Z;                           // f32 NHWC 4x64x64x128
constexpr size_t SZ_UPF    = (size_t)NNV * DHIN * DWIN * DC * 4;
constexpr size_t OFF_UPBF  = OFF_UPF + SZ_UPF;                       // bf16 copy
constexpr size_t SZ_UPBF   = (size_t)NNV * DHIN * DWIN * DC * 2;
constexpr size_t OFF_MID   = OFF_UPBF + SZ_UPBF;                     // bf16 NHWC 4x64x64x256
constexpr size_t SZ_MID    = (size_t)NNV * DHIN * DWIN * 2 * DC * 2;
// repacked bf16 weights, [col][k] so B-fragment K-pairs are contiguous dwords
constexpr size_t OFF_WKVRP = OFF_MID + SZ_MID;                       // [l][256][128]
constexpr size_t SZ_WKVRP  = (size_t)2 * 256 * 128 * 2;
constexpr size_t OFF_W1RP  = OFF_WKVRP + SZ_WKVRP;                   // [l][256][128]
constexpr size_t SZ_W1RP   = (size_t)2 * 256 * 128 * 2;
constexpr size_t OFF_W2RP  = OFF_W1RP + SZ_W1RP;                     // [l][128][256]
constexpr size_t SZ_W2RP   = (size_t)2 * 128 * 256 * 2;
constexpr size_t OFF_R1RP  = OFF_W2RP + SZ_W2RP;                     // [kk][256][128]
constexpr size_t SZ_R1RP   = (size_t)49 * 256 * 128 * 2;
constexpr size_t OFF_R2RP  = OFF_R1RP + SZ_R1RP;                     // [kk][128][256]
constexpr size_t SZ_R2RP   = (size_t)49 * 128 * 256 * 2;
constexpr size_t OFF_WQRP  = OFF_R2RP + SZ_R2RP;                     // [l][128][128]
constexpr size_t SZ_WQRP   = (size_t)2 * 128 * 128 * 2;
constexpr size_t OFF_WORP  = OFF_WQRP + SZ_WQRP;                     // [l][128][128]
constexpr size_t SZ_WORP   = (size_t)2 * 128 * 128 * 2;
constexpr size_t OFF_QBUF  = OFF_WORP + SZ_WORP;                     // f32 [nt][128]
constexpr size_t SZ_QBUF   = (size_t)NT * DC * 4;
constexpr size_t OFF_OBUF  = OFF_QBUF + SZ_QBUF;                     // bf16 [nt][128]
constexpr size_t SZ_OBUF   = (size_t)NT * DC * 2;
constexpr size_t OFF_YBF   = OFF_OBUF + SZ_OBUF;                     // bf16 [nt][128] (ln1 out)
constexpr size_t SZ_YBF    = (size_t)NT * DC * 2;

// ---- bf16 convert (round-to-nearest-even) ----
DEV unsigned short f2bf(float f) {
  unsigned u = __builtin_bit_cast(unsigned, f);
  unsigned r = u + 0x7FFFu + ((u >> 16) & 1u);
  return (unsigned short)(r >> 16);
}

// ---- WMMA fragment index maps (CDNA5 ISA 7.12.4, 16-bit 16x16x32) ----
DEV int a_kidx(int e, int hgrp) {
  int p = e >> 1, lo = e & 1;
  int base = (p < 4) ? ((hgrp ? 8 : 0) + 2 * p) : ((hgrp ? 24 : 16) + 2 * (p - 4));
  return base + lo;
}
DEV int b_kidx(int e, int hgrp) {
  int p = e >> 1, lo = e & 1;
  int base = (p < 4) ? ((hgrp ? 8 : 0) + 2 * p) : ((hgrp ? 24 : 16) + 2 * (p - 4));
  return base + lo;
}

// fragment loaders: K-pairs are contiguous 2xbf16 -> one aligned dword each
DEV v16u frag_a_row(const unsigned short* row, int hgrp) {
  v16u a;
#pragma unroll
  for (int p = 0; p < 8; ++p) {
    int k0 = a_kidx(2 * p, hgrp);
    unsigned wv = *(const unsigned*)(row + k0);
    a[2 * p] = (unsigned short)wv;
    a[2 * p + 1] = (unsigned short)(wv >> 16);
  }
  return a;
}
DEV v16u frag_a_row_cond(const unsigned short* row, int hgrp, bool inb) {
  v16u a;
#pragma unroll
  for (int p = 0; p < 8; ++p) {
    int k0 = a_kidx(2 * p, hgrp);
    unsigned wv = inb ? *(const unsigned*)(row + k0) : 0u;
    a[2 * p] = (unsigned short)wv;
    a[2 * p + 1] = (unsigned short)(wv >> 16);
  }
  return a;
}
DEV v16u frag_b_row(const unsigned short* row, int hgrp) {
  v16u b;
#pragma unroll
  for (int p = 0; p < 8; ++p) {
    int k0 = b_kidx(2 * p, hgrp);
    unsigned wv = *(const unsigned*)(row + k0);
    b[2 * p] = (unsigned short)wv;
    b[2 * p + 1] = (unsigned short)(wv >> 16);
  }
  return b;
}

DEV v8f wmma_bf16(v16u a, v16u b, v8f c) {
  return __builtin_amdgcn_wmma_f32_16x16x32_bf16(
      false, __builtin_bit_cast(v16bf, a),
      false, __builtin_bit_cast(v16bf, b),
      (short)0, c, false, false);
}

DEV v8f vzero8() { v8f z = {0.f,0.f,0.f,0.f,0.f,0.f,0.f,0.f}; return z; }
DEV float gelu_exact(float x) { return 0.5f * x * (1.f + erff(x * 0.70710678118654752f)); }

// =====================================================================
// Weight repack kernels: f32 [m][k][n] -> bf16 [m][n][k]
// =====================================================================
__global__ __launch_bounds__(256) void k_repack_mat(const float* __restrict__ in,
                                                    unsigned short* __restrict__ out,
                                                    int Mm, int K, int N) {
  int i = blockIdx.x * 256 + threadIdx.x;
  int total = Mm * K * N;
  if (i >= total) return;
  int m = i / (K * N), r = i - m * (K * N);
  int k = r / N, nn = r - k * N;
  out[((size_t)m * N + nn) * K + k] = f2bf(in[i]);
}
// conv weights f32 (OC, IC, 7, 7) -> bf16 [kk][oc][ic]
__global__ __launch_bounds__(256) void k_repack_conv(const float* __restrict__ in,
                                                     unsigned short* __restrict__ out,
                                                     int OC, int IC) {
  int i = blockIdx.x * 256 + threadIdx.x;
  int total = OC * IC * 49;
  if (i >= total) return;
  int oc = i / (IC * 49), r = i - oc * (IC * 49);
  int ic = r / 49, kk = r - ic * 49;
  out[((size_t)kk * OC + oc) * IC + ic] = f2bf(in[i]);
}

// =====================================================================
// Kernel 1: stride-2 2x2 conv as implicit GEMM (M=4096, N=128, K=512)
// =====================================================================
__global__ __launch_bounds__(32) void k_downconv(const float* __restrict__ fin,
                                                 const float* __restrict__ w,
                                                 const float* __restrict__ bias,
                                                 float* __restrict__ feats) {
  int lane = threadIdx.x, hgrp = lane >> 4, lm = lane & 15;
  int tm = blockIdx.x, tn = blockIdx.y;
  int oc = tn * 16 + lm;
  int mrow = tm * 16 + lm;
  int n = mrow >> 10, hh = (mrow >> 5) & 31, ww = mrow & 31;
  v8f acc = vzero8();
  for (int kb = 0; kb < 16; ++kb) {
    int khkw = kb >> 2, icb = (kb & 3) * 32;
    int kh = khkw >> 1, kw = khkw & 1;
    const float* src = fin + (size_t)n * DC * DHIN * DWIN
                       + (size_t)(2 * hh + kh) * DWIN + (2 * ww + kw);
    v16u a, bb;
#pragma unroll
    for (int e = 0; e < 16; ++e) {
      int ic = icb + a_kidx(e, hgrp);
      a[e] = f2bf(src[(size_t)ic * DHIN * DWIN]);
    }
#pragma unroll
    for (int e = 0; e < 16; ++e) {
      int ic = icb + b_kidx(e, hgrp);
      bb[e] = f2bf(w[((size_t)oc * DC + ic) * 4 + khkw]);
    }
    acc = wmma_bf16(a, bb, acc);
  }
  float bv = bias[oc];
#pragma unroll
  for (int i = 0; i < 8; ++i) {
    int m = tm * 16 + i + hgrp * 8;
    int nn = m >> 10, h2 = (m >> 5) & 31, w2 = m & 31;
    feats[(((size_t)(nn * DH + h2)) * DW + w2) * DC + oc] = acc[i] + bv;
  }
}

// =====================================================================
// Kernel 2: epipolar geometry + bilinear sample + depth embedding -> z (bf16)
// Stage 1: threads 0..95 compute per-sample xy/disp/pos-enc into LDS (once).
// Stage 2: 128 channel-threads consume them. (no 128x redundant sin/cos)
// =====================================================================
DEV void inv3(const float* k, float* o) {
  float d = k[0]*(k[4]*k[8]-k[5]*k[7]) - k[1]*(k[3]*k[8]-k[5]*k[6]) + k[2]*(k[3]*k[7]-k[4]*k[6]);
  float id = 1.f / d;
  o[0] = (k[4]*k[8]-k[5]*k[7])*id; o[1] = (k[2]*k[7]-k[1]*k[8])*id; o[2] = (k[1]*k[5]-k[2]*k[4])*id;
  o[3] = (k[5]*k[6]-k[3]*k[8])*id; o[4] = (k[0]*k[8]-k[2]*k[6])*id; o[5] = (k[2]*k[3]-k[0]*k[5])*id;
  o[6] = (k[3]*k[7]-k[4]*k[6])*id; o[7] = (k[1]*k[6]-k[0]*k[7])*id; o[8] = (k[0]*k[4]-k[1]*k[3])*id;
}

__global__ __launch_bounds__(128) void k_geom(const float* __restrict__ feats,
                                              const float* __restrict__ extr,
                                              const float* __restrict__ intr,
                                              const float* __restrict__ nearv,
                                              const float* __restrict__ farv,
                                              const float* __restrict__ depth_w,
                                              const float* __restrict__ depth_b,
                                              unsigned short* __restrict__ zbf) {
  __shared__ float pe_sh[DS][2 * DOCT];
  __shared__ float xsh[DS], ysh[DS];

  int nr = blockIdx.x;
  int n = nr >> 10, r = nr & 1023;
  int b = n >> 1, v = n & 1, vo = v ^ 1;
  const float* E  = extr + (size_t)(b * DV + v)  * 16;
  const float* Eo = extr + (size_t)(b * DV + vo) * 16;
  const float* K  = intr + (size_t)(b * DV + v)  * 9;
  const float* Ko = intr + (size_t)(b * DV + vo) * 9;
  float nearf = nearv[b * DV + v], farf = farv[b * DV + v];

  int t = threadIdx.x;
  if (t < DS) {
    float Ki[9], Koi[9];
    inv3(K, Ki); inv3(Ko, Koi);
    float Rw[9], tw[3];
    for (int i = 0; i < 3; ++i) {
      for (int j = 0; j < 3; ++j) Rw[i * 3 + j] = Eo[j * 4 + i];
      tw[i] = -(Eo[0 * 4 + i] * Eo[3] + Eo[1 * 4 + i] * Eo[7] + Eo[2 * 4 + i] * Eo[11]);
    }
    int xs = r & 31, ys = r >> 5;
    float px = (xs + 0.5f) / (float)DW, py = (ys + 0.5f) / (float)DH;
    float dc0 = Ki[0]*px + Ki[1]*py + Ki[2];
    float dc1 = Ki[3]*px + Ki[4]*py + Ki[5];
    float dc2 = Ki[6]*px + Ki[7]*py + Ki[8];
    float dx = E[0]*dc0 + E[1]*dc1 + E[2]*dc2;
    float dy = E[4]*dc0 + E[5]*dc1 + E[6]*dc2;
    float dz = E[8]*dc0 + E[9]*dc1 + E[10]*dc2;
    float inl = rsqrtf(dx*dx + dy*dy + dz*dz);
    dx *= inl; dy *= inl; dz *= inl;
    float ox = E[3], oy = E[7], oz = E[11];

    float uv[4];
    for (int t2 = 0; t2 < 2; ++t2) {
      float tt = t2 ? farf : nearf;
      float p0 = ox + tt * dx, p1 = oy + tt * dy, p2 = oz + tt * dz;
      float c0 = Rw[0]*p0 + Rw[1]*p1 + Rw[2]*p2 + tw[0];
      float c1 = Rw[3]*p0 + Rw[4]*p1 + Rw[5]*p2 + tw[1];
      float c2 = Rw[6]*p0 + Rw[7]*p1 + Rw[8]*p2 + tw[2];
      float zz = fmaxf(c2, FEPS);
      float q0 = c0 / zz, q1 = c1 / zz, q2 = c2 / zz;
      uv[t2 * 2 + 0] = Ko[0]*q0 + Ko[1]*q1 + Ko[2]*q2;
      uv[t2 * 2 + 1] = Ko[3]*q0 + Ko[4]*q1 + Ko[5]*q2;
    }
    float sgx = uv[2] - uv[0], sgy = uv[3] - uv[1];
    float tmin = 0.f, tmax = 1.f;
    for (int a = 0; a < 2; ++a) {
      float pa = a ? uv[1] : uv[0];
      float da = a ? sgy : sgx;
      bool par = fabsf(da) < FEPS;
      float das = par ? 1.f : da;
      float t1 = (0.f - pa) / das, t2 = (1.f - pa) / das;
      float lo = fminf(t1, t2), hi = fmaxf(t1, t2);
      bool inside = (pa >= 0.f) && (pa <= 1.f);
      if (par) { lo = inside ? 0.f : 1.f; hi = inside ? 1.f : 0.f; }
      tmin = fmaxf(tmin, lo); tmax = fminf(tmax, hi);
    }
    tmax = fmaxf(tmax, tmin);
    float usx = uv[0] + tmin * sgx, usy = uv[1] + tmin * sgy;
    float uex = uv[0] + tmax * sgx, uey = uv[1] + tmax * sgy;

    float oc0 = Rw[0]*ox + Rw[1]*oy + Rw[2]*oz + tw[0];
    float oc1 = Rw[3]*ox + Rw[4]*oy + Rw[5]*oz + tw[1];
    float oc2 = Rw[6]*ox + Rw[7]*oy + Rw[8]*oz + tw[2];
    float dc0o = Rw[0]*dx + Rw[1]*dy + Rw[2]*dz;
    float dc1o = Rw[3]*dx + Rw[4]*dy + Rw[5]*dz;
    float dc2o = Rw[6]*dx + Rw[7]*dy + Rw[8]*dz;

    float inv_near = 1.f / nearf, inv_far = 1.f / farf;
    float disp_den = 1.f / (inv_near - inv_far);

    int s = t;
    float frac = (s + 0.5f) / (float)DS;
    float xx = usx + frac * (uex - usx);
    float yy = usy + frac * (uey - usy);
    float c3x = Koi[0]*xx + Koi[1]*yy + Koi[2];
    float c3y = Koi[3]*xx + Koi[4]*yy + Koi[5];
    float c3z = Koi[6]*xx + Koi[7]*yy + Koi[8];
    float v1x = oc1*c3z - oc2*c3y, v1y = oc2*c3x - oc0*c3z, v1z = oc0*c3y - oc1*c3x;
    float v2x = c3y*dc2o - c3z*dc1o, v2y = c3z*dc0o - c3x*dc2o, v2z = c3x*dc1o - c3y*dc0o;
    float num = v1x*v2x + v1y*v2y + v1z*v2z;
    float den = fmaxf(v2x*v2x + v2y*v2y + v2z*v2z, FEPS);
    float depth = fminf(fmaxf(num / den, nearf), farf);
    float disp = 1.f - (1.f / depth - inv_far) * disp_den;
    float ang = disp * 6.283185307179586f;
#pragma unroll
    for (int k2 = 0; k2 < DOCT; ++k2) {
      pe_sh[s][k2] = sinf(ang);
      pe_sh[s][DOCT + k2] = cosf(ang);
      ang *= 2.f;
    }
    xsh[s] = xx; ysh[s] = yy;
  }
  __syncthreads();

  int c = t;
  const float* fb = feats + (size_t)(b * DV + vo) * DH * DW * DC;
  unsigned short* zout = zbf + ((size_t)nr * DS) * DC + c;
  float db = depth_b[c];

  for (int s = 0; s < DS; ++s) {
    float xx = xsh[s], yy = ysh[s];
    float pxs = xx * (float)DW - 0.5f, pys = yy * (float)DH - 0.5f;
    float x0f = floorf(pxs), y0f = floorf(pys);
    float wx = pxs - x0f, wy = pys - y0f;
    int x0 = min(max((int)x0f, 0), DW - 1);
    int y0 = min(max((int)y0f, 0), DH - 1);
    int x1 = min(x0 + 1, DW - 1);
    int y1 = min(y0 + 1, DH - 1);
    float f00 = fb[((size_t)(y0 * DW + x0)) * DC + c];
    float f01 = fb[((size_t)(y0 * DW + x1)) * DC + c];
    float f10 = fb[((size_t)(y1 * DW + x0)) * DC + c];
    float f11 = fb[((size_t)(y1 * DW + x1)) * DC + c];
    float samp = f00 * (1.f - wx) * (1.f - wy) + f01 * wx * (1.f - wy)
               + f10 * (1.f - wx) * wy + f11 * wx * wy;
    float emb = db;
#pragma unroll
    for (int k2 = 0; k2 < 2 * DOCT; ++k2) emb += pe_sh[s][k2] * depth_w[k2 * DC + c];
    zout[(size_t)s * DC] = f2bf(samp + emb);
  }
}

// =====================================================================
// Kernel: token LayerNorm -> bf16 (one token per 128-thread block)
// =====================================================================
__global__ __launch_bounds__(128) void k_ln_bf16(const float* __restrict__ x,
                                                 const float* __restrict__ g,
                                                 const float* __restrict__ bsh,
                                                 unsigned short* __restrict__ out,
                                                 int layer) {
  __shared__ float red[128];
  int t = threadIdx.x;
  int nr = blockIdx.x;
  float xv = x[(size_t)nr * DC + t];
  red[t] = xv;
  __syncthreads();
  for (int st = 64; st > 0; st >>= 1) { if (t < st) red[t] += red[t + st]; __syncthreads(); }
  float mean = red[0] * (1.f / 128.f);
  __syncthreads();
  float dv = xv - mean;
  red[t] = dv * dv;
  __syncthreads();
  for (int st = 64; st > 0; st >>= 1) { if (t < st) red[t] += red[t + st]; __syncthreads(); }
  float var = red[0] * (1.f / 128.f);
  out[(size_t)nr * DC + t] =
      f2bf(dv * rsqrtf(var + 1e-5f) * g[layer * DC + t] + bsh[layer * DC + t]);
}

// =====================================================================
// Batched token GEMMs (M=4096, K=128, N=128), one wave per 16x16 tile
// =====================================================================
__global__ __launch_bounds__(32) void k_gemm_q(const unsigned short* __restrict__ abf,
                                               const unsigned short* __restrict__ brp,
                                               float* __restrict__ outq) {
  int lane = threadIdx.x, hgrp = lane >> 4, lm = lane & 15;
  int tm = blockIdx.x, tn = blockIdx.y;
  int col = tn * 16 + lm;
  int mrow = tm * 16 + lm;
  const unsigned short* arow = abf + (size_t)mrow * DC;
  const unsigned short* brow = brp + (size_t)col * DC;
  v8f acc = vzero8();
#pragma unroll
  for (int kb = 0; kb < 4; ++kb) {
    v16u a = frag_a_row(arow + kb * 32, hgrp);
    v16u bb = frag_b_row(brow + kb * 32, hgrp);
    acc = wmma_bf16(a, bb, acc);
  }
#pragma unroll
  for (int i = 0; i < 8; ++i) {
    int m = tm * 16 + i + hgrp * 8;
    outq[(size_t)m * DC + col] = acc[i];
  }
}

__global__ __launch_bounds__(32) void k_gemm_o(const unsigned short* __restrict__ abf,
                                               const unsigned short* __restrict__ brp,
                                               const float* __restrict__ bias,
                                               float* __restrict__ x) {
  int lane = threadIdx.x, hgrp = lane >> 4, lm = lane & 15;
  int tm = blockIdx.x, tn = blockIdx.y;
  int col = tn * 16 + lm;
  int mrow = tm * 16 + lm;
  const unsigned short* arow = abf + (size_t)mrow * DC;
  const unsigned short* brow = brp + (size_t)col * DC;
  v8f acc = vzero8();
#pragma unroll
  for (int kb = 0; kb < 4; ++kb) {
    v16u a = frag_a_row(arow + kb * 32, hgrp);
    v16u bb = frag_b_row(brow + kb * 32, hgrp);
    acc = wmma_bf16(a, bb, acc);
  }
  float bv = bias[col];
#pragma unroll
  for (int i = 0; i < 8; ++i) {
    int m = tm * 16 + i + hgrp * 8;
    x[(size_t)m * DC + col] += acc[i] + bv;   // residual: x = attn(ln1(x)) + x
  }
}

// =====================================================================
// Kernel 3: attention core. block = 256 (8 waves), grid = 4096.
// z tile staged HBM->LDS via TDM (tensor_load_to_lds) when available.
// KV projection 96x128 @ 128x256 via WMMA; scores/softmax/AV VALU.
// =====================================================================
__global__ __launch_bounds__(256) void k_attn(const float* __restrict__ qbuf,
                                              const unsigned short* __restrict__ zbf,
                                              const unsigned short* __restrict__ wkvrp,
                                              unsigned short* __restrict__ obuf,
                                              int layer) {
  extern __shared__ char smem[];
  float* kv_sh = (float*)smem;                        // 96*256
  float* Q  = kv_sh + DS * 2 * DINNER;                // 128
  float* SC = Q + DC;                                 // 4*96
  unsigned short* z_sh = (unsigned short*)(SC + DHEADS * DS); // 96*128 (16B aligned)

  int t = threadIdx.x, lane = t & 31, wave = t >> 5;
  int hgrp = (lane >> 4), lm = lane & 15;
  int nr = blockIdx.x;

  if (t < DC) Q[t] = qbuf[(size_t)nr * DC + t];

#if USE_TDM
  if (wave == 0) {
    // Tensor DMA: copy 96x128 bf16 (24576 B) tile HBM -> LDS as a 1-D tensor.
    constexpr unsigned NEL = DS * DC;                 // 12288 elements, data_size=2B
    unsigned lds_off = (unsigned)(size_t)(void*)z_sh; // low 32 bits = LDS byte address
    unsigned long long ga = (unsigned long long)(const void*)(zbf + (size_t)nr * NEL);
    v4ui g0;
    g0[0] = 1u;                                       // count=1, user descriptor
    g0[1] = lds_off;                                  // lds_addr
    g0[2] = (unsigned)ga;                             // global_addr[31:0]
    g0[3] = (unsigned)((ga >> 32) & 0x01FFFFFFu) | 0x80000000u; // addr[56:32] | type=2
    unsigned long long q0 = (1ull << 16)              // data_size=1 (2 bytes)
                          | ((unsigned long long)(NEL & 0xFFFFu) << 48);   // tensor_dim0 lo
    unsigned long long q1 = (unsigned long long)(NEL >> 16)                // tensor_dim0 hi
                          | (1ull << 16)                                   // tensor_dim1=1
                          | ((unsigned long long)NEL << 48);               // tile_dim0
    unsigned long long q2 = 1ull                                           // tile_dim1=1
                          | ((unsigned long long)NEL << 32);               // tensor_dim0_stride lo32
    unsigned long long q3 = 0ull;
    v8si g1;
    g1[0] = (int)q0; g1[1] = (int)(q0 >> 32);
    g1[2] = (int)q1; g1[3] = (int)(q1 >> 32);
    g1[4] = (int)q2; g1[5] = (int)(q2 >> 32);
    g1[6] = (int)q3; g1[7] = (int)(q3 >> 32);
    v4si gz = {0, 0, 0, 0};
#if defined(__clang_major__) && __clang_major__ >= 23
    v8si gz8 = {0, 0, 0, 0, 0, 0, 0, 0};
    __builtin_amdgcn_tensor_load_to_lds(g0, g1, gz, gz, gz8, 0);
#else
    __builtin_amdgcn_tensor_load_to_lds(g0, g1, gz, gz, 0);
#endif
    __builtin_amdgcn_s_wait_tensorcnt(0);
  }
#else
  {
    const v4ui* src = (const v4ui*)(zbf + (size_t)nr * DS * DC);
    v4ui* dst = (v4ui*)z_sh;
    for (int i = t; i < DS * DC * 2 / 16; i += 256) dst[i] = src[i];
  }
#endif
  __syncthreads();
  // KV projection: (96x128) @ (128x256) via WMMA; B-frags cached over M tiles
  for (int tn = wave * 2; tn < wave * 2 + 2; ++tn) {
    int col = tn * 16 + lm;
    const unsigned short* wrow = wkvrp + ((size_t)layer * 2 * DINNER + col) * DC;
    v16u bfrag[4];
#pragma unroll
    for (int kb = 0; kb < 4; ++kb) bfrag[kb] = frag_b_row(wrow + kb * 32, hgrp);
    for (int tmi = 0; tmi < 6; ++tmi) {
      v8f acc = vzero8();
#pragma unroll
      for (int kb = 0; kb < 4; ++kb) {
        v16u a = frag_a_row(z_sh + (size_t)(tmi * 16 + lm) * DC + kb * 32, hgrp);
        acc = wmma_bf16(a, bfrag[kb], acc);
      }
#pragma unroll
      for (int i = 0; i < 8; ++i)
        kv_sh[(size_t)(tmi * 16 + i + hgrp * 8) * (2 * DINNER) + col] = acc[i];
    }
  }
  __syncthreads();
  // scores = q . k / sqrt(32)
  for (int idx = t; idx < DHEADS * DS; idx += 256) {
    int h = idx / DS, s = idx - h * DS;
    const float* kk = kv_sh + (size_t)s * (2 * DINNER) + h * DDH;
    const float* qq = Q + h * DDH;
    float a = 0.f;
#pragma unroll
    for (int d = 0; d < DDH; ++d) a += qq[d] * kk[d];
    SC[h * DS + s] = a * 0.17677669529663687f;
  }
  __syncthreads();
  if (t < DHEADS) {
    float mx = -1e30f;
    for (int s = 0; s < DS; ++s) mx = fmaxf(mx, SC[t * DS + s]);
    float sum = 0.f;
    for (int s = 0; s < DS; ++s) { float e = expf(SC[t * DS + s] - mx); SC[t * DS + s] = e; sum += e; }
    float inv = 1.f / sum;
    for (int s = 0; s < DS; ++s) SC[t * DS + s] *= inv;
  }
  __syncthreads();
  if (t < DINNER) {
    int h = t >> 5;
    float a = 0.f;
    for (int s = 0; s < DS; ++s) a += SC[h * DS + s] * kv_sh[(size_t)s * (2 * DINNER) + DINNER + t];
    obuf[(size_t)nr * DC + t] = f2bf(a);
  }
}

// =====================================================================
// Kernel 4: fused MLP layer (16 tokens per block). WMMA for 128->256->128.
// =====================================================================
__global__ __launch_bounds__(256) void k_mlp(float* __restrict__ x,
                                             const float* __restrict__ ln_g,
                                             const float* __restrict__ ln_b,
                                             const unsigned short* __restrict__ w1rp,
                                             const float* __restrict__ b1,
                                             const unsigned short* __restrict__ w2rp,
                                             const float* __restrict__ b2,
                                             int layer) {
  __shared__ float Xs[16 * DC];
  __shared__ float Yf[16 * DC];
  __shared__ unsigned short Ybf[16 * DC];
  __shared__ unsigned short Hbf[16 * DMLPD];
  __shared__ float mv[16], rs[16];

  int t = threadIdx.x, lane = t & 31, wave = t >> 5;
  int hgrp = lane >> 4, lm = lane & 15;
  int tok0 = blockIdx.x * 16;

  for (int i = t; i < 16 * DC; i += 256) Xs[i] = x[(size_t)tok0 * DC + i];
  __syncthreads();
  if (t < 16) {
    const float* xr = Xs + t * DC;
    float s = 0.f;
    for (int cc = 0; cc < DC; ++cc) s += xr[cc];
    float m = s * (1.f / 128.f), vv = 0.f;
    for (int cc = 0; cc < DC; ++cc) { float d = xr[cc] - m; vv += d * d; }
    mv[t] = m; rs[t] = rsqrtf(vv * (1.f / 128.f) + 1e-5f);
  }
  __syncthreads();
  for (int i = t; i < 16 * DC; i += 256) {
    int tok = i >> 7, cc = i & 127;
    float y = (Xs[i] - mv[tok]) * rs[tok] * ln_g[layer * DC + cc] + ln_b[layer * DC + cc];
    Yf[i] = y; Ybf[i] = f2bf(y);
  }
  __syncthreads();
  // GEMM1: (16x128) @ (128x256) + bias, gelu
  for (int tn = wave * 2; tn < wave * 2 + 2; ++tn) {
    int col = tn * 16 + lm;
    const unsigned short* wrow = w1rp + ((size_t)layer * DMLPD + col) * DC;
    v8f acc = vzero8();
#pragma unroll
    for (int kb = 0; kb < 4; ++kb) {
      v16u a = frag_a_row(Ybf + lm * DC + kb * 32, hgrp);
      v16u bb = frag_b_row(wrow + kb * 32, hgrp);
      acc = wmma_bf16(a, bb, acc);
    }
    float bias = b1[layer * DMLPD + col];
#pragma unroll
    for (int i = 0; i < 8; ++i) {
      int m = i + hgrp * 8;
      Hbf[m * DMLPD + col] = f2bf(gelu_exact(acc[i] + bias));
    }
  }
  __syncthreads();
  // GEMM2: (16x256) @ (256x128) + bias + y + x
  {
    int col = wave * 16 + lm;
    const unsigned short* wrow = w2rp + ((size_t)layer * DC + col) * DMLPD;
    v8f acc = vzero8();
#pragma unroll
    for (int kb = 0; kb < 8; ++kb) {
      v16u a = frag_a_row(Hbf + lm * DMLPD + kb * 32, hgrp);
      v16u bb = frag_b_row(wrow + kb * 32, hgrp);
      acc = wmma_bf16(a, bb, acc);
    }
    float bias = b2[layer * DC + col];
#pragma unroll
    for (int i = 0; i < 8; ++i) {
      int m = i + hgrp * 8;
      x[(size_t)(tok0 + m) * DC + col] = acc[i] + bias + Yf[m * DC + col] + Xs[m * DC + col];
    }
  }
}

// =====================================================================
// Kernel 5: 2x2 stride-2 transposed conv, output NHWC f32 + bf16
// =====================================================================
__global__ __launch_bounds__(32) void k_upconv(const float* __restrict__ x,
                                               const float* __restrict__ upw,
                                               const float* __restrict__ upb,
                                               float* __restrict__ upf,
                                               unsigned short* __restrict__ upbf) {
  int lane = threadIdx.x, hgrp = lane >> 4, lm = lane & 15;
  int tm = blockIdx.x, tn = blockIdx.y, khkw = blockIdx.z;
  int kh = khkw >> 1, kw = khkw & 1;
  int oc = tn * 16 + lm;
  int mrow = tm * 16 + lm;
  v8f acc = vzero8();
#pragma unroll
  for (int kb = 0; kb < 4; ++kb) {
    v16u a, bb;
#pragma unroll
    for (int e = 0; e < 16; ++e) a[e] = f2bf(x[(size_t)mrow * DC + kb * 32 + a_kidx(e, hgrp)]);
#pragma unroll
    for (int e = 0; e < 16; ++e) {
      int ic = kb * 32 + b_kidx(e, hgrp);
      bb[e] = f2bf(upw[((size_t)ic * DC + oc) * 4 + khkw]);
    }
    acc = wmma_bf16(a, bb, acc);
  }
  float bv = upb[oc];
#pragma unroll
  for (int i = 0; i < 8; ++i) {
    int m = tm * 16 + i + hgrp * 8;
    int n = m >> 10, hh = (m >> 5) & 31, ww = m & 31;
    size_t idx = (((size_t)(n * DHIN + (2 * hh + kh))) * DWIN + (2 * ww + kw)) * DC + oc;
    float val = acc[i] + bv;
    upf[idx] = val;
    upbf[idx] = f2bf(val);
  }
}

// =====================================================================
// Kernel 6: 7x7 conv C->2C + gelu (implicit GEMM, bf16 repacked weights)
// =====================================================================
__global__ __launch_bounds__(32) void k_ref1(const unsigned short* __restrict__ in_bf,
                                             const unsigned short* __restrict__ wrp,
                                             const float* __restrict__ bias,
                                             unsigned short* __restrict__ out_bf) {
  int lane = threadIdx.x, hgrp = lane >> 4, lm = lane & 15;
  int tm = blockIdx.x, tn = blockIdx.y;
  int oc = tn * 16 + lm;
  int mrow = tm * 16 + lm;
  int n = mrow >> 12, rem = mrow & 4095, yR = rem >> 6, xR = rem & 63;
  v8f acc = vzero8();
  for (int kk = 0; kk < 49; ++kk) {
    int kh = kk / 7, kw = kk - kh * 7;
    int yy = yR + kh - 3, xw = xR + kw - 3;
    bool inb = (yy >= 0) && (yy < DHIN) && (xw >= 0) && (xw < DWIN);
    const unsigned short* arow = in_bf + (((size_t)(n * DHIN + yy)) * DWIN + xw) * DC;
    const unsigned short* wrow = wrp + ((size_t)kk * (2 * DC) + oc) * DC;
    __builtin_prefetch(wrp + ((size_t)(kk + 1) * (2 * DC) + oc) * DC, 0, 1);
#pragma unroll
    for (int icb = 0; icb < 4; ++icb) {
      v16u a = frag_a_row_cond(arow + icb * 32, hgrp, inb);
      v16u bb = frag_b_row(wrow + icb * 32, hgrp);
      acc = wmma_bf16(a, bb, acc);
    }
  }
  float bv = bias[oc];
#pragma unroll
  for (int i = 0; i < 8; ++i) {
    int m = tm * 16 + i + hgrp * 8;
    int nn = m >> 12, rm = m & 4095, y2 = rm >> 6, x2 = rm & 63;
    out_bf[(((size_t)(nn * DHIN + y2)) * DWIN + x2) * (2 * DC) + oc] = f2bf(gelu_exact(acc[i] + bv));
  }
}

// =====================================================================
// Kernel 7: 7x7 conv 2C->C + bias + residual, write NCHW output
// =====================================================================
__global__ __launch_bounds__(32) void k_ref2(const unsigned short* __restrict__ in_bf,
                                             const unsigned short* __restrict__ wrp,
                                             const float* __restrict__ bias,
                                             const float* __restrict__ upf,
                                             float* __restrict__ out) {
  int lane = threadIdx.x, hgrp = lane >> 4, lm = lane & 15;
  int tm = blockIdx.x, tn = blockIdx.y;
  int oc = tn * 16 + lm;
  int mrow = tm * 16 + lm;
  int n = mrow >> 12, rem = mrow & 4095, yR = rem >> 6, xR = rem & 63;
  v8f acc = vzero8();
  for (int kk = 0; kk < 49; ++kk) {
    int kh = kk / 7, kw = kk - kh * 7;
    int yy = yR + kh - 3, xw = xR + kw - 3;
    bool inb = (yy >= 0) && (yy < DHIN) && (xw >= 0) && (xw < DWIN);
    const unsigned short* arow = in_bf + (((size_t)(n * DHIN + yy)) * DWIN + xw) * (2 * DC);
    const unsigned short* wrow = wrp + ((size_t)kk * DC + oc) * (2 * DC);
    __builtin_prefetch(wrp + ((size_t)(kk + 1) * DC + oc) * (2 * DC), 0, 1);
#pragma unroll
    for (int icb = 0; icb < 8; ++icb) {
      v16u a = frag_a_row_cond(arow + icb * 32, hgrp, inb);
      v16u bb = frag_b_row(wrow + icb * 32, hgrp);
      acc = wmma_bf16(a, bb, acc);
    }
  }
  float bv = bias[oc];
#pragma unroll
  for (int i = 0; i < 8; ++i) {
    int m = tm * 16 + i + hgrp * 8;
    int nn = m >> 12, rm = m & 4095, y2 = rm >> 6, x2 = rm & 63;
    float val = acc[i] + bv + upf[(((size_t)(nn * DHIN + y2)) * DWIN + x2) * DC + oc];
    out[(size_t)nn * DC * DHIN * DWIN + (size_t)oc * DHIN * DWIN + (size_t)y2 * DWIN + x2] = val;
  }
}

// =====================================================================
extern "C" void kernel_launch(void* const* d_in, const int* in_sizes, int n_in,
                              void* d_out, int out_size, void* d_ws, size_t ws_size,
                              hipStream_t stream) {
  const float* features = (const float*)d_in[0];
  const float* extr     = (const float*)d_in[1];
  const float* intr     = (const float*)d_in[2];
  const float* nearv    = (const float*)d_in[3];
  const float* farv     = (const float*)d_in[4];
  const float* down_w   = (const float*)d_in[5];
  const float* down_b   = (const float*)d_in[6];
  const float* depth_w  = (const float*)d_in[7];
  const float* depth_b  = (const float*)d_in[8];
  const float* ln1_g    = (const float*)d_in[9];
  const float* ln1_b    = (const float*)d_in[10];
  const float* wq       = (const float*)d_in[11];
  const float* wkv      = (const float*)d_in[12];
  const float* wo       = (const float*)d_in[13];
  const float* bo       = (const float*)d_in[14];
  const float* ln2_g    = (const float*)d_in[15];
  const float* ln2_b    = (const float*)d_in[16];
  const float* w1       = (const float*)d_in[17];
  const float* b1       = (const float*)d_in[18];
  const float* w2       = (const float*)d_in[19];
  const float* b2       = (const float*)d_in[20];
  const float* up_w     = (const float*)d_in[21];
  const float* up_b     = (const float*)d_in[22];
  const float* ref1_w   = (const float*)d_in[23];
  const float* ref1_b   = (const float*)d_in[24];
  const float* ref2_w   = (const float*)d_in[25];
  const float* ref2_b   = (const float*)d_in[26];

  char* ws = (char*)d_ws;
  float*          feats = (float*)(ws + OFF_FEATS);   // also x tokens (aliased)
  unsigned short* zbf   = (unsigned short*)(ws + OFF_Z);
  float*          upf   = (float*)(ws + OFF_UPF);
  unsigned short* upbf  = (unsigned short*)(ws + OFF_UPBF);
  unsigned short* midbf = (unsigned short*)(ws + OFF_MID);
  unsigned short* wkvrp = (unsigned short*)(ws + OFF_WKVRP);
  unsigned short* w1rp  = (unsigned short*)(ws + OFF_W1RP);
  unsigned short* w2rp  = (unsigned short*)(ws + OFF_W2RP);
  unsigned short* r1rp  = (unsigned short*)(ws + OFF_R1RP);
  unsigned short* r2rp  = (unsigned short*)(ws + OFF_R2RP);
  unsigned short* wqrp  = (unsigned short*)(ws + OFF_WQRP);
  unsigned short* worp  = (unsigned short*)(ws + OFF_WORP);
  float*          qbuf  = (float*)(ws + OFF_QBUF);
  unsigned short* obuf  = (unsigned short*)(ws + OFF_OBUF);
  unsigned short* ybf   = (unsigned short*)(ws + OFF_YBF);

  // weight repacks (f32 -> bf16, [col][k] layouts)
  k_repack_mat<<<(2 * 128 * 256 + 255) / 256, 256, 0, stream>>>(wkv, wkvrp, 2, 128, 256);
  k_repack_mat<<<(2 * 128 * 256 + 255) / 256, 256, 0, stream>>>(w1, w1rp, 2, 128, 256);
  k_repack_mat<<<(2 * 256 * 128 + 255) / 256, 256, 0, stream>>>(w2, w2rp, 2, 256, 128);
  k_repack_mat<<<(2 * 128 * 128 + 255) / 256, 256, 0, stream>>>(wq, wqrp, 2, 128, 128);
  k_repack_mat<<<(2 * 128 * 128 + 255) / 256, 256, 0, stream>>>(wo, worp, 2, 128, 128);
  k_repack_conv<<<(256 * 128 * 49 + 255) / 256, 256, 0, stream>>>(ref1_w, r1rp, 256, 128);
  k_repack_conv<<<(128 * 256 * 49 + 255) / 256, 256, 0, stream>>>(ref2_w, r2rp, 128, 256);

  k_downconv<<<dim3(NT / 16, DC / 16), 32, 0, stream>>>(features, down_w, down_b, feats);
  k_geom<<<NT, DC, 0, stream>>>(feats, extr, intr, nearv, farv, depth_w, depth_b, zbf);

  constexpr size_t ATTN_SMEM =
      (size_t)(DS * 2 * DINNER + DC + DHEADS * DS) * 4 + (size_t)DS * DC * 2;
  for (int l = 0; l < 2; ++l) {
    k_ln_bf16<<<NT, DC, 0, stream>>>(feats, ln1_g, ln1_b, ybf, l);
    k_gemm_q<<<dim3(NT / 16, DC / 16), 32, 0, stream>>>(
        ybf, wqrp + (size_t)l * DINNER * DC, qbuf);
    k_attn<<<NT, 256, ATTN_SMEM, stream>>>(qbuf, zbf, wkvrp, obuf, l);
    k_gemm_o<<<dim3(NT / 16, DC / 16), 32, 0, stream>>>(
        obuf, worp + (size_t)l * DINNER * DC, bo + (size_t)l * DC, feats);
    k_mlp<<<NT / 16, 256, 0, stream>>>(feats, ln2_g, ln2_b, w1rp, b1, w2rp, b2, l);
  }

  k_upconv<<<dim3(NT / 16, DC / 16, 4), 32, 0, stream>>>(feats, up_w, up_b, upf, upbf);
  k_ref1<<<dim3((NNV * DHIN * DWIN) / 16, (2 * DC) / 16), 32, 0, stream>>>(upbf, r1rp, ref1_b, midbf);
  k_ref2<<<dim3((NNV * DHIN * DWIN) / 16, DC / 16), 32, 0, stream>>>(midbf, r2rp, ref2_b, upf, (float*)d_out);
}
